// GlobalAttentionLite_7017976562181
// MI455X (gfx1250) — matrix operand order repaired and checked by reading, not compiled, not run on hardware
//
#include <hip/hip_runtime.h>
#include <hip/hip_bf16.h>

// ---------------------------------------------------------------------------
// GlobalAttentionLite on MI455X (gfx1250): bf16 WMMA everywhere, flash-style
// fused softmax-attention, BN/SiLU epilogues fused into the WMMA kernels.
// GEMM kernels compute a 16x64 strip per wave: one A fragment is reused
// across 4 B fragments / 4 accumulators per k-step (4 independent WMMAs
// cover the XDL RAW latency and cut A-operand traffic 4x).
// ---------------------------------------------------------------------------

typedef __attribute__((ext_vector_type(16))) __bf16 v16bf;
typedef __attribute__((ext_vector_type(8)))  __bf16 v8bf;
typedef __attribute__((ext_vector_type(8)))  float  v8f;

#define B_   8
#define C_   256
#define H_   128
#define W_   128
#define N_   (H_ * W_)        // 16384
#define M_   ((H_/4)*(W_/4))  // 1024
#define AC_  64
#define VC_  128
#define EPS_ 1e-5f

// ----------------------------- helpers -------------------------------------

__device__ __forceinline__ __bf16 to_bf16(float f) {
  unsigned u = __builtin_bit_cast(unsigned, f);
  u += 0x7FFFu + ((u >> 16) & 1u);          // round-to-nearest-even
  unsigned short h = (unsigned short)(u >> 16);
  return __builtin_bit_cast(__bf16, h);
}

// A-matrix 16x32 bf16 fragment (CDNA5 layout: lane holds row lane%16,
// K = half*8..half*8+7 in elems 0..7 and 16+half*8.. in elems 8..15).
__device__ __forceinline__ v16bf load_a_frag(const __bf16* A, int lda,
                                             int row0, int k0, int lane) {
  const int r    = row0 + (lane & 15);
  const int half = lane >> 4;
  const __bf16* rp = A + (long)r * lda + k0;
  v8bf lo = *(const v8bf*)(rp + half * 8);
  v8bf hi = *(const v8bf*)(rp + 16 + half * 8);
  v16bf out;
#pragma unroll
  for (int i = 0; i < 8; ++i) { out[i] = lo[i]; out[i + 8] = hi[i]; }
  return out;
}

// B-matrix 32x16 bf16 fragment from column-major storage (per column the
// K dimension is contiguous).  Lane holds column lane%16, K = half*8..half*8+7
// in elems 0..7 and 16+half*8.. in elems 8..15.
__device__ __forceinline__ v16bf load_b_frag(const __bf16* Bm, int ldb,
                                             int col0, int k0, int lane) {
  const int c    = col0 + (lane & 15);
  const int half = lane >> 4;
  const __bf16* cp = Bm + (long)c * ldb + k0;
  v8bf lo = *(const v8bf*)(cp + half * 8);
  v8bf hi = *(const v8bf*)(cp + 16 + half * 8);
  v16bf out;
#pragma unroll
  for (int i = 0; i < 8; ++i) { out[i] = lo[i]; out[i + 8] = hi[i]; }
  return out;
}

__device__ __forceinline__ v8f wmma_bf16(v16bf a, v16bf b, v8f c) {
  return __builtin_amdgcn_wmma_f32_16x16x32_bf16(
      /*neg_a=*/false, a, /*neg_b=*/false, b,
      /*c_mod=*/(short)0, c, /*reuse_a=*/false, /*reuse_b=*/false);
}

// 16 x (16*NB) output strip: one A fragment reused across NB B fragments.
template <int NB>
__device__ __forceinline__ void wmma_strip(const __bf16* A, int lda, int row0,
                                           const __bf16* Bm, int ldb, int col0,
                                           int K, int lane, v8f* acc) {
#pragma unroll
  for (int j = 0; j < NB; ++j) acc[j] = {};
  for (int k = 0; k < K; k += 32) {
    v16bf a = load_a_frag(A, lda, row0, k, lane);
#pragma unroll
    for (int j = 0; j < NB; ++j)
      acc[j] = wmma_bf16(a, load_b_frag(Bm, ldb, col0 + 16 * j, k, lane),
                         acc[j]);
  }
}

// ----------------------- data-prep kernels ----------------------------------

__global__ void cast_bf16_kernel(const float* __restrict__ src,
                                 __bf16* __restrict__ dst, int n) {
  int i = blockIdx.x * blockDim.x + threadIdx.x;
  if (i < n) dst[i] = to_bf16(src[i]);
}

// x [B][C][N] f32  ->  xb [B][N][C] bf16  (B-operand layout for Q GEMM)
__global__ void xpose_kernel(const float* __restrict__ x,
                             __bf16* __restrict__ xb) {
  long i = (long)blockIdx.x * blockDim.x + threadIdx.x;
  int n = (int)(i % N_);
  long t = i / N_;
  int c = (int)(t % C_);
  int b = (int)(t / C_);
  xb[((long)b * N_ + n) * C_ + c] = to_bf16(x[i]);
}

// 4x4 average pool: x [B][C][H][W] -> xkv [B][M][C] bf16
__global__ void pool_kernel(const float* __restrict__ x,
                            __bf16* __restrict__ xkv) {
  int i = blockIdx.x * blockDim.x + threadIdx.x;   // B*M*C threads
  int c = i % C_;
  int t = i / C_;
  int m = t % M_;
  int b = t / M_;
  int mh = m >> 5, mw = m & 31;
  const float* p = x + ((long)b * C_ + c) * N_ + (mh * 4) * W_ + mw * 4;
  float s = 0.f;
#pragma unroll
  for (int dy = 0; dy < 4; ++dy)
#pragma unroll
    for (int dx = 0; dx < 4; ++dx) s += p[dy * W_ + dx];
  xkv[((long)b * M_ + m) * C_ + c] = to_bf16(s * (1.f / 16.f));
}

// ----------------------- projection GEMMs (Q, K, V) -------------------------

// Q = wq[64,256] x x[256,N]; scale folded in; store qb [B][N][AC]
__global__ void __launch_bounds__(128)
gemm_q_kernel(const __bf16* __restrict__ wqb, const __bf16* __restrict__ xb,
              __bf16* __restrict__ qb) {
  int lane = threadIdx.x & 31;
  int tile = blockIdx.x * 4 + (threadIdx.x >> 5);
  int nt = tile % (N_ / 64); tile /= (N_ / 64);
  int at = tile % (AC_ / 16);
  int b  = tile / (AC_ / 16);
  const __bf16* Bm = xb + (long)b * N_ * C_;
  v8f acc[4];
  wmma_strip<4>(wqb, C_, at * 16, Bm, C_, nt * 64, C_, lane, acc);
  const float scale = 0.125f;                 // 64^-0.5
  int col = lane & 15, half = lane >> 4;
  __bf16* q = qb + (long)b * N_ * AC_;
#pragma unroll
  for (int j = 0; j < 4; ++j)
#pragma unroll
    for (int r = 0; r < 8; ++r) {
      int a = at * 16 + r + 8 * half;
      int n = nt * 64 + j * 16 + col;
      q[(long)n * AC_ + a] = to_bf16(acc[j][r] * scale);
    }
}

// K = wk[64,256] x xkv[256,M]; store kb [B][M][AC]
__global__ void __launch_bounds__(128)
gemm_k_kernel(const __bf16* __restrict__ wkb, const __bf16* __restrict__ xkvb,
              __bf16* __restrict__ kb) {
  int lane = threadIdx.x & 31;
  int tile = blockIdx.x * 4 + (threadIdx.x >> 5);
  int mt = tile % (M_ / 64); tile /= (M_ / 64);
  int at = tile % (AC_ / 16);
  int b  = tile / (AC_ / 16);
  const __bf16* Bm = xkvb + (long)b * M_ * C_;
  v8f acc[4];
  wmma_strip<4>(wkb, C_, at * 16, Bm, C_, mt * 64, C_, lane, acc);
  int col = lane & 15, half = lane >> 4;
  __bf16* k = kb + (long)b * M_ * AC_;
#pragma unroll
  for (int j = 0; j < 4; ++j)
#pragma unroll
    for (int r = 0; r < 8; ++r) {
      int a = at * 16 + r + 8 * half;
      int m = mt * 64 + j * 16 + col;
      k[(long)m * AC_ + a] = to_bf16(acc[j][r]);
    }
}

// V = wv[128,256] x xkv[256,M]; store vb [B][VC][M]  (m contiguous per v-row)
__global__ void __launch_bounds__(128)
gemm_v_kernel(const __bf16* __restrict__ wvb, const __bf16* __restrict__ xkvb,
              __bf16* __restrict__ vb) {
  int lane = threadIdx.x & 31;
  int tile = blockIdx.x * 4 + (threadIdx.x >> 5);
  int mt = tile % (M_ / 64); tile /= (M_ / 64);
  int vt = tile % (VC_ / 16);
  int b  = tile / (VC_ / 16);
  const __bf16* Bm = xkvb + (long)b * M_ * C_;
  v8f acc[4];
  wmma_strip<4>(wvb, C_, vt * 16, Bm, C_, mt * 64, C_, lane, acc);
  int col = lane & 15, half = lane >> 4;
  __bf16* v = vb + (long)b * VC_ * M_;
#pragma unroll
  for (int j = 0; j < 4; ++j)
#pragma unroll
    for (int r = 0; r < 8; ++r) {
      int vr = vt * 16 + r + 8 * half;
      int m  = mt * 64 + j * 16 + col;
      v[(long)vr * M_ + m] = to_bf16(acc[j][r]);
    }
}

// ----------------------- flash attention ------------------------------------
// One wave per 16-query tile. S = Q.K (WMMA), online softmax across M chunks
// of 32 keys, P restaged via LDS into A-layout, y += P.V (WMMA, VC=128).
__global__ void __launch_bounds__(32)
flash_kernel(const __bf16* __restrict__ qb, const __bf16* __restrict__ kb,
             const __bf16* __restrict__ vb, __bf16* __restrict__ yb) {
  __shared__ float pstage[16 * 32];
  int lane = threadIdx.x;
  int tile = blockIdx.x;                    // b * (N/16) tiles
  int nt = tile % (N_ / 16);
  int b  = tile / (N_ / 16);
  const __bf16* Q = qb + (long)b * N_ * AC_;
  const __bf16* K = kb + (long)b * M_ * AC_;
  const __bf16* V = vb + (long)b * VC_ * M_;

  v16bf aq0 = load_a_frag(Q, AC_, nt * 16, 0,  lane);
  v16bf aq1 = load_a_frag(Q, AC_, nt * 16, 32, lane);

  v8f acc[8];
#pragma unroll
  for (int vc = 0; vc < 8; ++vc) acc[vc] = {};
  float mrun[8], lrun[8];
#pragma unroll
  for (int r = 0; r < 8; ++r) { mrun[r] = -1e30f; lrun[r] = 0.f; }

  int col = lane & 15, half = lane >> 4;

  for (int m0 = 0; m0 < M_; m0 += 32) {
    // ---- scores: two 16x16 tiles (keys m0..15, m0+16..31) ----
    v8f s0 = {}, s1 = {};
    s0 = wmma_bf16(aq0, load_b_frag(K, AC_, m0,      0,  lane), s0);
    s0 = wmma_bf16(aq1, load_b_frag(K, AC_, m0,      32, lane), s0);
    s1 = wmma_bf16(aq0, load_b_frag(K, AC_, m0 + 16, 0,  lane), s1);
    s1 = wmma_bf16(aq1, load_b_frag(K, AC_, m0 + 16, 32, lane), s1);

    if (m0 + 32 < M_)                        // global_prefetch_b8 on gfx1250
      __builtin_prefetch(K + (long)(m0 + 32) * AC_, 0, 0);

    // ---- online softmax (row reductions across the 16-lane half-group) ----
#pragma unroll
    for (int r = 0; r < 8; ++r) {
      float mx = fmaxf(s0[r], s1[r]);
#pragma unroll
      for (int off = 1; off < 16; off <<= 1)
        mx = fmaxf(mx, __shfl_xor(mx, off, 32));
      float mnew  = fmaxf(mrun[r], mx);
      float alpha = __expf(mrun[r] - mnew);
      float e0 = __expf(s0[r] - mnew);
      float e1 = __expf(s1[r] - mnew);
      float rs = e0 + e1;
#pragma unroll
      for (int off = 1; off < 16; off <<= 1)
        rs += __shfl_xor(rs, off, 32);
      lrun[r] = lrun[r] * alpha + rs;
      mrun[r] = mnew;
#pragma unroll
      for (int vc = 0; vc < 8; ++vc) acc[vc][r] *= alpha;
      int row = r + 8 * half;                // C-layout row of this lane
      pstage[row * 32 + col]      = e0;
      pstage[row * 32 + 16 + col] = e1;
    }
    __syncthreads();   // single-wave WG: barrier is S_NOP, keeps the DS wait

    // ---- restage P into A-layout (16 queries x 32 keys) ----
    v16bf pa;
    {
      const float* pr = &pstage[(lane & 15) * 32];
#pragma unroll
      for (int j = 0; j < 8; ++j) {
        pa[j]     = to_bf16(pr[half * 8 + j]);
        pa[j + 8] = to_bf16(pr[16 + half * 8 + j]);
      }
    }
    // ---- y += P.V over all 8 v-chunks (VC = 128) ----
#pragma unroll
    for (int vc = 0; vc < 8; ++vc)
      acc[vc] = wmma_bf16(pa, load_b_frag(V, M_, vc * 16, m0, lane), acc[vc]);
    __syncthreads();
  }

  __bf16* Y = yb + (long)b * N_ * VC_;
#pragma unroll
  for (int r = 0; r < 8; ++r) {
    float inv = 1.f / lrun[r];
    int n = nt * 16 + r + 8 * half;
#pragma unroll
    for (int vc = 0; vc < 8; ++vc)
      Y[(long)n * VC_ + vc * 16 + col] = to_bf16(acc[vc][r] * inv);
  }
}

// ----------------------- proj + BN + SiLU -----------------------------------

__global__ void __launch_bounds__(128)
gemm_proj_kernel(const __bf16* __restrict__ wpb, const __bf16* __restrict__ yb,
                 float* __restrict__ ya,
                 const float* __restrict__ g,  const float* __restrict__ bb,
                 const float* __restrict__ mm, const float* __restrict__ vv) {
  int lane = threadIdx.x & 31;
  int tile = blockIdx.x * 4 + (threadIdx.x >> 5);
  int nt = tile % (N_ / 64); tile /= (N_ / 64);
  int ct = tile % (C_ / 16);
  int b  = tile / (C_ / 16);
  const __bf16* Bm = yb + (long)b * N_ * VC_;
  v8f acc[4];
  wmma_strip<4>(wpb, VC_, ct * 16, Bm, VC_, nt * 64, VC_, lane, acc);
  int col = lane & 15, half = lane >> 4;
#pragma unroll
  for (int r = 0; r < 8; ++r) {
    int c = ct * 16 + r + 8 * half;
    float inv  = g[c] * rsqrtf(vv[c] + EPS_);
    float beta = bb[c] - mm[c] * inv;
#pragma unroll
    for (int j = 0; j < 4; ++j) {
      float val = acc[j][r] * inv + beta;
      val = val / (1.f + __expf(-val));                  // SiLU
      ya[((long)b * C_ + c) * N_ + nt * 64 + j * 16 + col] = val;
    }
  }
}

// ----------------------- depthwise 3x3 --------------------------------------

__global__ void dwconv_kernel(const float* __restrict__ ya,
                              const float* __restrict__ dww,
                              __bf16* __restrict__ zb) {
  long i = (long)blockIdx.x * blockDim.x + threadIdx.x;  // B*C*N threads
  int n = (int)(i % N_);
  long t = i / N_;
  int c = (int)(t % C_);
  int b = (int)(t / C_);
  int h = n >> 7, w = n & 127;
  const float* base = ya + ((long)b * C_ + c) * N_;
  const float* wt   = dww + c * 9;
  float s = 0.f;
#pragma unroll
  for (int dy = -1; dy <= 1; ++dy)
#pragma unroll
    for (int dx = -1; dx <= 1; ++dx) {
      int hh = h + dy, ww = w + dx;
      if (hh >= 0 && hh < H_ && ww >= 0 && ww < W_)
        s += wt[(dy + 1) * 3 + (dx + 1)] * base[hh * W_ + ww];
    }
  zb[((long)b * N_ + n) * C_ + c] = to_bf16(s);
}

// ----------------------- pointwise + BN + residual + SiLU -------------------

__global__ void __launch_bounds__(128)
gemm_pw_kernel(const __bf16* __restrict__ pwb, const __bf16* __restrict__ zb,
               const float* __restrict__ x, float* __restrict__ out,
               const float* __restrict__ g,  const float* __restrict__ bb,
               const float* __restrict__ mm, const float* __restrict__ vv) {
  int lane = threadIdx.x & 31;
  int tile = blockIdx.x * 4 + (threadIdx.x >> 5);
  int nt = tile % (N_ / 64); tile /= (N_ / 64);
  int ct = tile % (C_ / 16);
  int b  = tile / (C_ / 16);
  const __bf16* Bm = zb + (long)b * N_ * C_;
  v8f acc[4];
  wmma_strip<4>(pwb, C_, ct * 16, Bm, C_, nt * 64, C_, lane, acc);
  int col = lane & 15, half = lane >> 4;
#pragma unroll
  for (int r = 0; r < 8; ++r) {
    int c = ct * 16 + r + 8 * half;
    float inv  = g[c] * rsqrtf(vv[c] + EPS_);
    float beta = bb[c] - mm[c] * inv;
#pragma unroll
    for (int j = 0; j < 4; ++j) {
      float val = acc[j][r] * inv + beta;
      long  idx = ((long)b * C_ + c) * N_ + nt * 64 + j * 16 + col;
      out[idx] = x[idx] + val / (1.f + __expf(-val));    // x + SiLU(z)
    }
  }
}

// ----------------------- launch ---------------------------------------------

extern "C" void kernel_launch(void* const* d_in, const int* in_sizes, int n_in,
                              void* d_out, int out_size, void* d_ws, size_t ws_size,
                              hipStream_t stream) {
  const float* x     = (const float*)d_in[0];
  const float* wq    = (const float*)d_in[1];
  const float* wk    = (const float*)d_in[2];
  const float* wv    = (const float*)d_in[3];
  const float* wproj = (const float*)d_in[4];
  const float* bn_g  = (const float*)d_in[5];
  const float* bn_b  = (const float*)d_in[6];
  const float* bn_m  = (const float*)d_in[7];
  const float* bn_v  = (const float*)d_in[8];
  const float* dw_w  = (const float*)d_in[9];
  const float* pw_w  = (const float*)d_in[10];
  const float* mbn_g = (const float*)d_in[11];
  const float* mbn_b = (const float*)d_in[12];
  const float* mbn_m = (const float*)d_in[13];
  const float* mbn_v = (const float*)d_in[14];
  float* out = (float*)d_out;

  char*  ws  = (char*)d_ws;
  size_t off = 0;
  auto carve = [&](size_t bytes) -> void* {
    void* p = ws + off;
    off += (bytes + 255) & ~(size_t)255;
    return p;
  };
  __bf16* wqb   = (__bf16*)carve((size_t)AC_ * C_  * 2);
  __bf16* wkb   = (__bf16*)carve((size_t)AC_ * C_  * 2);
  __bf16* wvb   = (__bf16*)carve((size_t)VC_ * C_  * 2);
  __bf16* wpb   = (__bf16*)carve((size_t)C_  * VC_ * 2);
  __bf16* pwb   = (__bf16*)carve((size_t)C_  * C_  * 2);
  __bf16* xb    = (__bf16*)carve((size_t)B_ * N_ * C_  * 2);
  __bf16* xkvb  = (__bf16*)carve((size_t)B_ * M_ * C_  * 2);
  __bf16* qb    = (__bf16*)carve((size_t)B_ * N_ * AC_ * 2);
  __bf16* kb    = (__bf16*)carve((size_t)B_ * M_ * AC_ * 2);
  __bf16* vb    = (__bf16*)carve((size_t)B_ * VC_ * M_ * 2);
  __bf16* yb    = (__bf16*)carve((size_t)B_ * N_ * VC_ * 2);
  float*  ya    = (float*) carve((size_t)B_ * C_ * N_  * 4);
  __bf16* zb    = (__bf16*)carve((size_t)B_ * N_ * C_  * 2);
  (void)ws_size; (void)n_in; (void)in_sizes; (void)out_size;

  // 1) weights to bf16
  cast_bf16_kernel<<<(AC_*C_ + 255)/256, 256, 0, stream>>>(wq, wqb, AC_*C_);
  cast_bf16_kernel<<<(AC_*C_ + 255)/256, 256, 0, stream>>>(wk, wkb, AC_*C_);
  cast_bf16_kernel<<<(VC_*C_ + 255)/256, 256, 0, stream>>>(wv, wvb, VC_*C_);
  cast_bf16_kernel<<<(C_*VC_ + 255)/256, 256, 0, stream>>>(wproj, wpb, C_*VC_);
  cast_bf16_kernel<<<(C_*C_  + 255)/256, 256, 0, stream>>>(pw_w, pwb, C_*C_);

  // 2) activations: transpose + pool
  xpose_kernel<<<(B_*(long)C_*N_)/256, 256, 0, stream>>>(x, xb);
  pool_kernel <<<(B_*M_*C_)/256,       256, 0, stream>>>(x, xkvb);

  // 3) Q/K/V projections (WMMA, 16x64 strips)
  gemm_q_kernel<<<(B_*(AC_/16)*(N_/64))/4, 128, 0, stream>>>(wqb, xb, qb);
  gemm_k_kernel<<<(B_*(AC_/16)*(M_/64))/4, 128, 0, stream>>>(wkb, xkvb, kb);
  gemm_v_kernel<<<(B_*(VC_/16)*(M_/64))/4, 128, 0, stream>>>(wvb, xkvb, vb);

  // 4) flash attention (WMMA + online softmax)
  flash_kernel<<<B_*(N_/16), 32, 0, stream>>>(qb, kb, vb, yb);

  // 5) projection + BN + SiLU  (WMMA, 16x64 strips)
  gemm_proj_kernel<<<(B_*(C_/16)*(N_/64))/4, 128, 0, stream>>>(
      wpb, yb, ya, bn_g, bn_b, bn_m, bn_v);

  // 6) depthwise 3x3
  dwconv_kernel<<<(B_*(long)C_*N_)/256, 256, 0, stream>>>(ya, dw_w, zb);

  // 7) pointwise + BN + residual + SiLU  (WMMA, 16x64 strips)
  gemm_pw_kernel<<<(B_*(C_/16)*(N_/64))/4, 128, 0, stream>>>(
      pwb, zb, x, out, mbn_g, mbn_b, mbn_m, mbn_v);
}
